// ThreadGNN_52931176956203
// MI455X (gfx1250) — matrix operand order repaired and checked by reading, not compiled, not run on hardware
//
#include <hip/hip_runtime.h>
#include <hip/hip_bf16.h>

// bf16 storage type = unsigned short (bit pattern); vector types only for WMMA.
typedef __attribute__((ext_vector_type(16))) __bf16 v16bf;
typedef __attribute__((ext_vector_type(8)))  __bf16 v8bf;
typedef __attribute__((ext_vector_type(8)))  float  v8f;

#define HID  256
#define NREL 6

typedef unsigned short bfr;  // raw bf16 bits

// -------- gfx1250 async global->LDS support (guarded) ----------------------
#if defined(__gfx1250__) && __has_builtin(__builtin_amdgcn_global_load_async_to_lds_b128)
#define HAS_ASYNC 1
#define AS1q __attribute__((address_space(1)))
#define AS3q __attribute__((address_space(3)))
typedef int v4i_vs __attribute__((vector_size(16)));  // matches builtin param pointee
#if __has_builtin(__builtin_amdgcn_s_wait_asynccnt)
#define WAIT_ASYNC0() __builtin_amdgcn_s_wait_asynccnt(0)
#else
#define WAIT_ASYNC0() asm volatile("s_wait_asynccnt 0x0" ::: "memory")
#endif
__device__ __forceinline__ void async_cp16(const bfr* g, bfr* l) {
  // 16B per lane, global -> LDS, tracked by ASYNCcnt
  __builtin_amdgcn_global_load_async_to_lds_b128((AS1q v4i_vs*)g, (AS3q v4i_vs*)l, 0, 0);
}
#else
#define HAS_ASYNC 0
#endif

__device__ __forceinline__ bfr f2bf(float f) {
  union { float f; unsigned u; } c; c.f = f;
  unsigned u = c.u + 0x7FFFu + ((c.u >> 16) & 1u);   // round-to-nearest-even
  return (bfr)(u >> 16);
}
__device__ __forceinline__ float bf2f(bfr h) {
  union { unsigned u; float f; } c; c.u = ((unsigned)h) << 16;
  return c.f;
}

// ---------------------------------------------------------------------------
// small helper kernels
// ---------------------------------------------------------------------------
__global__ void k_f32_to_bf16(const float* __restrict__ src, bfr* __restrict__ dst, int n) {
  int i = blockIdx.x * blockDim.x + threadIdx.x;
  if (i < n) dst[i] = f2bf(src[i]);
}

__global__ void k_embed(const int* __restrict__ x_type, const int* __restrict__ x_sub,
                        const float* __restrict__ type_emb, const float* __restrict__ sub_emb,
                        bfr* __restrict__ H, int n) {
  long long i = (long long)blockIdx.x * blockDim.x + threadIdx.x;
  if (i >= (long long)n * HID) return;
  int node = (int)(i >> 8), col = (int)(i & 255);
  float v = (col < 128) ? type_emb[x_type[node] * 128 + col]
                        : sub_emb[x_sub[node] * 128 + (col - 128)];
  H[i] = f2bf(v);
}

__global__ void k_degree(const int* __restrict__ edges, float* __restrict__ deg, int E) {
  int i = blockIdx.x * blockDim.x + threadIdx.x;
  if (i >= NREL * E) return;
  int r = i / E, e = i - r * E;
  int tgt = edges[r * 2 * E + E + e];
  atomicAdd(&deg[tgt], 1.0f);
}

// Acc[tgt,:] += T[src,:]  (64 threads per edge, float4 per thread)
__global__ void k_scatter(const int* __restrict__ edges_r, const float* __restrict__ T,
                          float* __restrict__ Acc, int E) {
  long long tid = (long long)blockIdx.x * blockDim.x + threadIdx.x;
  int e = (int)(tid >> 6);
  if (e >= E) return;
  int c = (int)(tid & 63) * 4;
  int src = edges_r[e];
  int tgt = edges_r[E + e];
  const float4 v = *(const float4*)(T + (size_t)src * HID + c);
  float* dst = Acc + (size_t)tgt * HID + c;
  atomicAdd(dst + 0, v.x); atomicAdd(dst + 1, v.y);
  atomicAdd(dst + 2, v.z); atomicAdd(dst + 3, v.w);
}

__global__ void k_pool_accum(const bfr* __restrict__ H, const int* __restrict__ bid,
                             float* __restrict__ pooled, float* __restrict__ cnt, int n) {
  long long tid = (long long)blockIdx.x * blockDim.x + threadIdx.x;
  int node = (int)(tid >> 6);
  if (node >= n) return;
  int c = (int)(tid & 63) * 4;
  int g = bid[node];
  const bfr* hp = H + (size_t)node * HID + c;
  float* dst = pooled + (size_t)g * HID + c;
  atomicAdd(dst + 0, bf2f(hp[0]));
  atomicAdd(dst + 1, bf2f(hp[1]));
  atomicAdd(dst + 2, bf2f(hp[2]));
  atomicAdd(dst + 3, bf2f(hp[3]));
  if (c == 0) atomicAdd(&cnt[g], 1.0f);
}

__global__ void k_pool_finish(const float* __restrict__ pooled, const float* __restrict__ cnt,
                              bfr* __restrict__ out, int total) {
  int i = blockIdx.x * blockDim.x + threadIdx.x;
  if (i >= total) return;
  int g = i >> 8;
  out[i] = f2bf(pooled[i] / fmaxf(cnt[g], 1.0f));
}

// ---------------------------------------------------------------------------
// WMMA GEMM:  C(MxHID) = A(MxHID,bf16) @ B(HIDxHID,bf16 row-major), K=N=256
// MODE 0: Cf = gemm
// MODE 1: Cb = relu(gemm + Msg/max(deg,1) + bias)   (RGCN combine, bf16 out)
// MODE 2: Cb = relu(gemm + bias)                    (bf16 out)
// MODE 3: Cf = gemm + bias                          (f32 out)
// ---------------------------------------------------------------------------
template <int MODE>
__device__ __forceinline__ void c_store(int row, int col, float v,
                                        float* __restrict__ Cf, bfr* __restrict__ Cb,
                                        const float* __restrict__ Msg,
                                        const float* __restrict__ deg,
                                        const float* __restrict__ bias) {
  if constexpr (MODE == 0) {
    Cf[(size_t)row * HID + col] = v;
  } else if constexpr (MODE == 1) {
    float d = fmaxf(deg[row], 1.0f);
    v = v + Msg[(size_t)row * HID + col] / d + bias[col];
    Cb[(size_t)row * HID + col] = f2bf(fmaxf(v, 0.0f));
  } else if constexpr (MODE == 2) {
    Cb[(size_t)row * HID + col] = f2bf(fmaxf(v + bias[col], 0.0f));
  } else {
    Cf[(size_t)row * HID + col] = v + bias[col];
  }
}

template <int MODE>
__global__ __launch_bounds__(256) void k_wmma_gemm(
    const bfr* __restrict__ A, const bfr* __restrict__ B, int M,
    float* __restrict__ Cf, bfr* __restrict__ Cb,
    const float* __restrict__ Msg, const float* __restrict__ deg,
    const float* __restrict__ bias) {
  constexpr int BM = 128, BN = 128, BK = 32;
  constexpr int NSTEP = HID / BK;  // 8
  constexpr int LDA = BK + 8;      // 40 elems -> 80B row stride (16B aligned)
  constexpr int LDB = BN + 8;      // 136 elems -> 272B row stride (16B aligned)

  const int tid  = threadIdx.x;
  const int lane = tid & 31;
  const int wave = tid >> 5;      // 0..7
  const int wm   = wave >> 1;     // 0..3  (rows of 32)
  const int wn   = wave & 1;      // 0..1  (cols of 64)
  const int m0   = blockIdx.y * BM;
  const int n0   = blockIdx.x * BN;

  v8f acc[2][4];
#pragma unroll
  for (int i = 0; i < 2; ++i)
#pragma unroll
    for (int j = 0; j < 4; ++j) {
      v8f z = {0.f, 0.f, 0.f, 0.f, 0.f, 0.f, 0.f, 0.f};
      acc[i][j] = z;
    }

  const int arow = tid >> 1, aseg = (tid & 1) * 16;   // 2 thr/row, 16 elems (32B) each
  const int brow = tid >> 3, bseg = (tid & 7) * 16;   // 8 thr/row
  const int khalf = lane >> 4;    // which K-half this lane holds (A frag)
  const int am    = lane & 15;    // A row within 16

#if HAS_ASYNC
  // ---- double-buffered async global->LDS pipeline ----
  __shared__ bfr As[2][BM * LDA];
  __shared__ bfr Bs[2][BK * LDB];

  auto stage = [&](int k0, int buf) {
    if (m0 + arow < M) {
      const bfr* gp = A + (size_t)(m0 + arow) * HID + k0 + aseg;
      bfr* lp = &As[buf][arow * LDA + aseg];
      async_cp16(gp, lp);
      async_cp16(gp + 8, lp + 8);
    } else {
      uint4 z = {0u, 0u, 0u, 0u};
      *(uint4*)(void*)&As[buf][arow * LDA + aseg]     = z;
      *(uint4*)(void*)&As[buf][arow * LDA + aseg + 8] = z;
    }
    const bfr* gp = B + (size_t)(k0 + brow) * HID + n0 + bseg;
    bfr* lp = &Bs[buf][brow * LDB + bseg];
    async_cp16(gp, lp);
    async_cp16(gp + 8, lp + 8);
  };

  stage(0, 0);
  for (int s = 0; s < NSTEP; ++s) {
    WAIT_ASYNC0();       // this wave's in-flight tile is in LDS
    __syncthreads();     // every wave's tile is in LDS; prev compute done
    if (s + 1 < NSTEP) stage((s + 1) * BK, (s + 1) & 1);
    const bfr* Asb = As[s & 1];
    const bfr* Bsb = Bs[s & 1];

    v16bf af[2], bfv[4];
#pragma unroll
    for (int mt = 0; mt < 2; ++mt) {
      const bfr* ap = &Asb[(wm * 32 + mt * 16 + am) * LDA];
      ((v8bf*)&af[mt])[0] = *(const v8bf*)(const void*)(ap + khalf * 8);
      ((v8bf*)&af[mt])[1] = *(const v8bf*)(const void*)(ap + 16 + khalf * 8);
    }
#pragma unroll
    for (int nt = 0; nt < 4; ++nt) {
      const bfr* bp = &Bsb[wn * 64 + nt * 16 + (lane & 15)];
#pragma unroll
      for (int s = 0; s < 16; ++s) {
        int kk = (s < 8) ? khalf * 8 + s : 16 + khalf * 8 + (s - 8);
        bfv[nt][s] = ((const __bf16*)(const void*)bp)[kk * LDB];
      }
    }
#pragma unroll
    for (int mt = 0; mt < 2; ++mt)
#pragma unroll
      for (int nt = 0; nt < 4; ++nt)
        acc[mt][nt] = __builtin_amdgcn_wmma_f32_16x16x32_bf16(
            false, af[mt], false, bfv[nt], (short)0, acc[mt][nt], false, false);
  }
#else
  // ---- fallback: synchronous register-staged single buffer ----
  __shared__ bfr As[BM * LDA];
  __shared__ bfr Bs[BK * LDB];
  for (int s = 0; s < NSTEP; ++s) {
    int k0 = s * BK;
    uint4 a0 = {0u, 0u, 0u, 0u}, a1 = {0u, 0u, 0u, 0u};
    if (m0 + arow < M) {
      const bfr* gp = A + (size_t)(m0 + arow) * HID + k0 + aseg;
      a0 = *(const uint4*)(const void*)gp;
      a1 = *(const uint4*)(const void*)(gp + 8);
    }
    *(uint4*)(void*)&As[arow * LDA + aseg]     = a0;
    *(uint4*)(void*)&As[arow * LDA + aseg + 8] = a1;
    {
      const bfr* gp = B + (size_t)(k0 + brow) * HID + n0 + bseg;
      *(uint4*)(void*)&Bs[brow * LDB + bseg]     = *(const uint4*)(const void*)gp;
      *(uint4*)(void*)&Bs[brow * LDB + bseg + 8] = *(const uint4*)(const void*)(gp + 8);
    }
    __syncthreads();
    v16bf af[2], bfv[4];
#pragma unroll
    for (int mt = 0; mt < 2; ++mt) {
      const bfr* ap = &As[(wm * 32 + mt * 16 + am) * LDA];
      ((v8bf*)&af[mt])[0] = *(const v8bf*)(const void*)(ap + khalf * 8);
      ((v8bf*)&af[mt])[1] = *(const v8bf*)(const void*)(ap + 16 + khalf * 8);
    }
#pragma unroll
    for (int nt = 0; nt < 4; ++nt) {
      const bfr* bp = &Bs[wn * 64 + nt * 16 + (lane & 15)];
#pragma unroll
      for (int s = 0; s < 16; ++s) {
        int kk = (s < 8) ? khalf * 8 + s : 16 + khalf * 8 + (s - 8);
        bfv[nt][s] = ((const __bf16*)(const void*)bp)[kk * LDB];
      }
    }
#pragma unroll
    for (int mt = 0; mt < 2; ++mt)
#pragma unroll
      for (int nt = 0; nt < 4; ++nt)
        acc[mt][nt] = __builtin_amdgcn_wmma_f32_16x16x32_bf16(
            false, af[mt], false, bfv[nt], (short)0, acc[mt][nt], false, false);
    __syncthreads();
  }
#endif

  // ---- epilogue: frag layout: VGPR i, lanes 0-15: M=i,N=lane; 16-31: M=8+i ----
  const int lrow = (lane >> 4) * 8;
  const int lcol = lane & 15;
  const int rbase = m0 + wm * 32 + lrow;
  const int cbase = n0 + wn * 64 + lcol;
  if (m0 + BM <= M) {  // full tile: branch-free stores
#pragma unroll
    for (int mt = 0; mt < 2; ++mt)
#pragma unroll
      for (int nt = 0; nt < 4; ++nt)
#pragma unroll
        for (int i = 0; i < 8; ++i)
          c_store<MODE>(rbase + mt * 16 + i, cbase + nt * 16, acc[mt][nt][i],
                        Cf, Cb, Msg, deg, bias);
  } else {
#pragma unroll
    for (int mt = 0; mt < 2; ++mt)
#pragma unroll
      for (int nt = 0; nt < 4; ++nt)
#pragma unroll
        for (int i = 0; i < 8; ++i) {
          int row = rbase + mt * 16 + i;
          if (row < M)
            c_store<MODE>(row, cbase + nt * 16, acc[mt][nt][i],
                          Cf, Cb, Msg, deg, bias);
        }
  }
}

// ---------------------------------------------------------------------------
extern "C" void kernel_launch(void* const* d_in, const int* in_sizes, int n_in,
                              void* d_out, int out_size, void* d_ws, size_t ws_size,
                              hipStream_t stream) {
  (void)n_in; (void)ws_size;
  const int*   x_type    = (const int*)d_in[0];
  const int*   x_sub     = (const int*)d_in[1];
  const int*   edges     = (const int*)d_in[2];
  const int*   batch_ids = (const int*)d_in[3];
  // d_in[4] = n_graphs scalar (unused on host; derived from out_size)
  const float* type_emb  = (const float*)d_in[5];
  const float* sub_emb   = (const float*)d_in[6];
  const float* W_rel0    = (const float*)d_in[7];
  const float* W_self0   = (const float*)d_in[8];
  const float* b0        = (const float*)d_in[9];
  const float* W_rel1    = (const float*)d_in[10];
  const float* W_self1   = (const float*)d_in[11];
  const float* b1        = (const float*)d_in[12];
  const float* pW1       = (const float*)d_in[13];
  const float* pb1       = (const float*)d_in[14];
  const float* pW2       = (const float*)d_in[15];
  const float* pb2       = (const float*)d_in[16];

  const int N = in_sizes[0];
  const int E = in_sizes[2] / (NREL * 2);
  const int G = out_size / HID;

  // workspace carve-up (≈310 MB)
  char* w = (char*)d_ws;
  auto take = [&](size_t bytes) {
    void* p = (void*)w;
    w += (bytes + 255) & ~(size_t)255;
    return p;
  };
  bfr*   HbfA   = (bfr*)  take((size_t)N * HID * 2);
  bfr*   HbfB   = (bfr*)  take((size_t)N * HID * 2);
  float* Acc    = (float*)take((size_t)N * HID * 4);
  float* T      = (float*)take((size_t)N * HID * 4);
  float* deg    = (float*)take((size_t)N * 4);
  bfr*   Wbf    = (bfr*)  take((size_t)16 * HID * HID * 2);
  float* pooled = (float*)take((size_t)G * HID * 4);
  float* cnt    = (float*)take((size_t)G * 4);
  bfr*   poolbf = (bfr*)  take((size_t)G * HID * 2);
  bfr*   hidbf  = (bfr*)  take((size_t)G * HID * 2);

  const int WSZ = HID * HID;  // 65536
  auto cvt = [&](const float* s, bfr* d, int n) {
    k_f32_to_bf16<<<(n + 255) / 256, 256, 0, stream>>>(s, d, n);
  };
  cvt(W_rel0,  Wbf + 0 * WSZ,  NREL * WSZ);
  cvt(W_self0, Wbf + 6 * WSZ,  WSZ);
  cvt(W_rel1,  Wbf + 7 * WSZ,  NREL * WSZ);
  cvt(W_self1, Wbf + 13 * WSZ, WSZ);
  cvt(pW1,     Wbf + 14 * WSZ, WSZ);
  cvt(pW2,     Wbf + 15 * WSZ, WSZ);

  // initial node features (bf16)
  k_embed<<<(unsigned)(((size_t)N * HID + 255) / 256), 256, 0, stream>>>(
      x_type, x_sub, type_emb, sub_emb, HbfA, N);

  // degrees (same for both layers)
  (void)hipMemsetAsync(deg, 0, (size_t)N * 4, stream);
  k_degree<<<(NREL * E + 255) / 256, 256, 0, stream>>>(edges, deg, E);

  dim3 ggrid(HID / 128, (N + 127) / 128);
  dim3 sgrid((unsigned)(((size_t)E * 64 + 255) / 256));

  // ---- layer 0 ----
  (void)hipMemsetAsync(Acc, 0, (size_t)N * HID * 4, stream);
  for (int r = 0; r < NREL; ++r) {
    k_wmma_gemm<0><<<ggrid, 256, 0, stream>>>(HbfA, Wbf + r * WSZ, N, T, nullptr,
                                              nullptr, nullptr, nullptr);
    k_scatter<<<sgrid, 256, 0, stream>>>(edges + r * 2 * E, T, Acc, E);
  }
  k_wmma_gemm<1><<<ggrid, 256, 0, stream>>>(HbfA, Wbf + 6 * WSZ, N, nullptr, HbfB,
                                            Acc, deg, b0);

  // ---- layer 1 ----
  (void)hipMemsetAsync(Acc, 0, (size_t)N * HID * 4, stream);
  for (int r = 0; r < NREL; ++r) {
    k_wmma_gemm<0><<<ggrid, 256, 0, stream>>>(HbfB, Wbf + (7 + r) * WSZ, N, T, nullptr,
                                              nullptr, nullptr, nullptr);
    k_scatter<<<sgrid, 256, 0, stream>>>(edges + r * 2 * E, T, Acc, E);
  }
  k_wmma_gemm<1><<<ggrid, 256, 0, stream>>>(HbfB, Wbf + 13 * WSZ, N, nullptr, HbfA,
                                            Acc, deg, b1);

  // ---- mean pool ----
  (void)hipMemsetAsync(pooled, 0, (size_t)G * HID * 4, stream);
  (void)hipMemsetAsync(cnt, 0, (size_t)G * 4, stream);
  k_pool_accum<<<(unsigned)(((size_t)N * 64 + 255) / 256), 256, 0, stream>>>(
      HbfA, batch_ids, pooled, cnt, N);
  k_pool_finish<<<(G * HID + 255) / 256, 256, 0, stream>>>(pooled, cnt, poolbf, G * HID);

  // ---- MLP head ----
  dim3 mgrid(HID / 128, (G + 127) / 128);
  k_wmma_gemm<2><<<mgrid, 256, 0, stream>>>(poolbf, Wbf + 14 * WSZ, G, nullptr, hidbf,
                                            nullptr, nullptr, pb1);
  k_wmma_gemm<3><<<mgrid, 256, 0, stream>>>(hidbf, Wbf + 15 * WSZ, G, (float*)d_out,
                                            nullptr, nullptr, nullptr, pb2);
}
